// SparseAttentionEngine_11252814316100
// MI455X (gfx1250) — matrix operand order repaired and checked by reading, not compiled, not run on hardware
//
#include <hip/hip_runtime.h>

// ---------- types ----------
typedef __attribute__((__ext_vector_type__(16))) __bf16 v16bf;
typedef __attribute__((__ext_vector_type__(8)))  float  v8f;
typedef __attribute__((__ext_vector_type__(4)))  unsigned v4u;
typedef __attribute__((__ext_vector_type__(8)))  unsigned v8u;

union BFrag {                    // one WMMA bf16 operand (8 VGPRs)
    v16bf v;
    unsigned short u[16];
    uint4 q[2];
};
union U4 {
    uint4 v;
    unsigned short s[8];
};

__device__ __forceinline__ unsigned short f2bf(float f) {
    unsigned u = __float_as_uint(f);
    u += 0x7fffu + ((u >> 16) & 1u);          // round-to-nearest-even
    return (unsigned short)(u >> 16);
}
__device__ __forceinline__ float bf2f(unsigned short h) {
    return __uint_as_float(((unsigned)h) << 16);
}
__device__ __forceinline__ unsigned lds_off(const void* p) {
    // generic pointer to LDS: low 32 bits are the LDS byte offset
    return (unsigned)(unsigned long long)p;
}

// async global->LDS copy of 16 bytes (ASYNCcnt-tracked, no VGPR data path)
__device__ __forceinline__ void async_copy_b128(unsigned lds_addr,
                                                const void* gaddr) {
    asm volatile("global_load_async_to_lds_b128 %0, %1, off" ::"v"(lds_addr),
                 "v"((unsigned long long)gaddr)
                 : "memory");
}
__device__ __forceinline__ void wait_async0() {
    asm volatile("s_wait_asynccnt 0x0" ::: "memory");
}

// ---------- constants ----------
#define NTOK 4096      // B*S
#define DMODEL 1024
#define NHEAD 4
#define HDIM 256
#define THRESH 0.15f
#define NEGINF (-1e9f)

// ---------- fp32 -> bf16 ----------
__global__ void cvt_kernel(const float* __restrict__ in,
                           unsigned short* __restrict__ out, int n) {
    int i = blockIdx.x * blockDim.x + threadIdx.x;
    int stride = gridDim.x * blockDim.x;
    for (; i < n; i += stride) out[i] = f2bf(in[i]);
}

// ---------- generic tiled WMMA GEMM: out = op(A@W + bias) ----------
// A [M,K] bf16 row-major, W [K,N] bf16 row-major.
// MODE 0: bf16 out, scaled       (QKV projections; Q uses scale=1/16)
// MODE 1: bf16 out, ReLU         (sparsity-predictor hidden layer)
// MODE 2: fp32 out, * maskmul[row] (output projection + query scatter-mask)
template <int MODE>
__global__ __launch_bounds__(128) void gemm_kernel(
    const unsigned short* __restrict__ A,
    const unsigned short* __restrict__ W,
    const float* __restrict__ bias,
    void* __restrict__ outp,
    const float* __restrict__ maskmul,
    int M, int N, int K, float scale) {
    __shared__ __align__(16) unsigned short As[64][40];   // 64x32 tile, padded
    __shared__ __align__(16) unsigned short Bt[16][40];   // 32x16 tile, transposed

    const int tid  = threadIdx.x;
    const int wave = tid >> 5;
    const int lane = tid & 31;
    const int l15  = lane & 15;
    const int half = lane >> 4;
    const int mb   = blockIdx.x * 64;
    const int nb   = blockIdx.y * 16;

    v8f c = {};

    for (int kt = 0; kt < K; kt += 32) {
        // stage A tile (64x32 bf16 = 256 x b128) via async global->LDS DMA
#pragma unroll
        for (int i = 0; i < 2; ++i) {
            int l  = tid + i * 128;
            int r  = l >> 2;
            int c8 = (l & 3) * 8;
            async_copy_b128(lds_off(&As[r][c8]),
                            A + (size_t)(mb + r) * K + kt + c8);
        }
        // stage W tile transposed (32x16 bf16) through VGPRs
        if (tid < 64) {
            int r  = tid >> 1;
            int c8 = (tid & 1) * 8;
            U4 vv;
            vv.v = *(const uint4*)(W + (size_t)(kt + r) * N + nb + c8);
#pragma unroll
            for (int e = 0; e < 8; ++e) Bt[c8 + e][r] = vv.s[e];
        }
        if (kt + 32 < K)   // next A tile into L2 (global_prefetch_b8)
            __builtin_prefetch(A + (size_t)(mb + (tid >> 2)) * K + kt + 32, 0, 3);
        wait_async0();
        __syncthreads();

        BFrag af, bf;
        const int row = wave * 16 + l15;
        af.q[0] = *(const uint4*)&As[row][half * 8];
        af.q[1] = *(const uint4*)&As[row][16 + half * 8];
        bf.q[0] = *(const uint4*)&Bt[l15][half * 8];
        bf.q[1] = *(const uint4*)&Bt[l15][16 + half * 8];
        c = __builtin_amdgcn_wmma_f32_16x16x32_bf16(false, af.v, false, bf.v,
                                                    (short)0, c, false, false);
        __syncthreads();
    }

    const float bn = bias[nb + l15];
#pragma unroll
    for (int j = 0; j < 8; ++j) {
        const int m   = j + half * 8;
        const int row = mb + wave * 16 + m;
        float val     = c[j] + bn;
        if constexpr (MODE == 0) {
            ((unsigned short*)outp)[(size_t)row * N + nb + l15] = f2bf(val * scale);
        } else if constexpr (MODE == 1) {
            ((unsigned short*)outp)[(size_t)row * N + nb + l15] =
                f2bf(val > 0.f ? val : 0.f);
        } else {
            ((float*)outp)[(size_t)row * N + nb + l15] = val * maskmul[row];
        }
    }
}

// ---------- sparsity scores -> key bias / query mask ----------
__global__ void score_kernel(const unsigned short* __restrict__ hbf,
                             const float* __restrict__ w2,
                             const float* __restrict__ b2,
                             float* __restrict__ maskbias,
                             float* __restrict__ maskmul, int n) {
    int t = blockIdx.x * blockDim.x + threadIdx.x;
    if (t >= n) return;
    const unsigned short* h = hbf + (size_t)t * 256;
    float s = b2[0];
#pragma unroll 8
    for (int j = 0; j < 256; ++j) s += bf2f(h[j]) * w2[j];
    float sig  = 1.f / (1.f + __expf(-s));
    bool  act  = sig > THRESH;
    maskbias[t] = act ? 0.f : NEGINF;
    maskmul[t]  = act ? 1.f : 0.f;
}

// ---------- flash attention: 4 waves/block, 16 queries/wave, 1 head/block.y ----
__global__ __launch_bounds__(128) void attn_kernel(
    const unsigned short* __restrict__ Qp,   // [N,1024] bf16 (pre-scaled)
    const unsigned short* __restrict__ Kp,
    const unsigned short* __restrict__ Vp,
    const float* __restrict__ maskbias,      // [N] 0 / -1e9 per key
    unsigned short* __restrict__ Ctx,        // [N,1024] bf16
    int N) {
    __shared__ __align__(16) unsigned short Ks[32][272];     // 32 keys x 256 dims
    __shared__ __align__(16) unsigned short Vt[256][40];     // transposed V tile
    __shared__ __align__(16) unsigned short Pt[4][16][40];   // per-wave P tile

    const int tid   = threadIdx.x;
    const int wave  = tid >> 5;
    const int lane  = tid & 31;
    const int l15   = lane & 15;
    const int half  = lane >> 4;
    const int doff  = blockIdx.y * HDIM;
    const int qbase = blockIdx.x * 64 + wave * 16;

    // Tensor-DMA descriptor group 1 (loop-invariant): 2D tile 256x32 of 2-byte
    // elements, tensor_dim0_stride=1024, LDS padding 32B per 512B row so the
    // TDM writes Ks[32][272] directly.
    v8u tg1;
    tg1[0] = (1u << 16)        // data_size = 2 bytes
           | (1u << 20)        // pad_enable
           | (6u << 22)        // pad_interval: 128 dwords (= one 512B row)
           | (7u << 25);       // pad_amount: 8 dwords (= 32B -> 272-short rows)
    tg1[1] = ((unsigned)DMODEL & 0xFFFFu) << 16;        // tensor_dim0 = 1024
    tg1[2] = ((unsigned)NTOK & 0xFFFFu) << 16;          // tensor_dim1 = 4096
    tg1[3] = ((unsigned)HDIM & 0xFFFFu) << 16;          // tile_dim0 = 256
    tg1[4] = 32u;                                       // tile_dim1 = 32 rows
    tg1[5] = (unsigned)DMODEL;                          // dim0_stride = 1024
    tg1[6] = 0u;
    tg1[7] = 0u;
    const unsigned ldsKs = lds_off(&Ks[0][0]);

    // Q fragments (16 rows x 256 dims, 8 K-tiles of 32), kept in VGPRs
    BFrag qf[8];
    const size_t qrow = (size_t)(qbase + l15) * DMODEL + doff;
#pragma unroll
    for (int t = 0; t < 8; ++t) {
        qf[t].q[0] = *(const uint4*)(Qp + qrow + t * 32 + half * 8);
        qf[t].q[1] = *(const uint4*)(Qp + qrow + t * 32 + 16 + half * 8);
    }

    v8f acc[16];
#pragma unroll
    for (int t = 0; t < 16; ++t) acc[t] = (v8f){};
    float mrow[8], lrow[8];
#pragma unroll
    for (int j = 0; j < 8; ++j) { mrow[j] = -3.0e38f; lrow[j] = 0.f; }

    for (int kb = 0; kb < N; kb += 32) {
        // K tile via Tensor Data Mover (wave 0 issues; TDM ignores EXEC, so
        // the guard is a wave-uniform branch)
        if (wave == 0) {
            unsigned long long ga =
                (unsigned long long)(Kp + (size_t)kb * DMODEL + doff);
            v4u tg0;
            tg0.x = 1u;                                  // count = 1 descriptor
            tg0.y = ldsKs;                               // lds_addr
            tg0.z = (unsigned)ga;                        // global_addr[31:0]
            tg0.w = (unsigned)(ga >> 32) & 0x01FFFFFFu;  // global_addr[56:32]
            tg0.w |= 0x80000000u;                        // type = 2 (image)
            asm volatile("tensor_load_to_lds %0, %1" ::"s"(tg0), "s"(tg1)
                         : "memory");
            __builtin_amdgcn_s_wait_tensorcnt(0);
        }
        // V tile staged transposed through VGPRs (all 128 threads)
#pragma unroll
        for (int i = 0; i < 8; ++i) {
            int l  = tid + i * 128;
            int r  = l >> 5;
            int c8 = (l & 31) * 8;
            U4 vv;
            vv.v = *(const uint4*)(Vp + (size_t)(kb + r) * DMODEL + doff + c8);
#pragma unroll
            for (int e = 0; e < 8; ++e) Vt[c8 + e][r] = vv.s[e];
        }
        __syncthreads();

        const float mb0 = maskbias[kb + l15];
        const float mb1 = maskbias[kb + 16 + l15];

        // S = Q . K^T   (16 queries x 32 keys), K-dim = 256 in 8 steps
        v8f s0 = {}, s1 = {};
#pragma unroll
        for (int t = 0; t < 8; ++t) {
            BFrag b0, b1;
            b0.q[0] = *(const uint4*)&Ks[l15][t * 32 + half * 8];
            b0.q[1] = *(const uint4*)&Ks[l15][t * 32 + 16 + half * 8];
            b1.q[0] = *(const uint4*)&Ks[16 + l15][t * 32 + half * 8];
            b1.q[1] = *(const uint4*)&Ks[16 + l15][t * 32 + 16 + half * 8];
            s0 = __builtin_amdgcn_wmma_f32_16x16x32_bf16(false, qf[t].v, false,
                                                         b0.v, (short)0, s0,
                                                         false, false);
            s1 = __builtin_amdgcn_wmma_f32_16x16x32_bf16(false, qf[t].v, false,
                                                         b1.v, (short)0, s1,
                                                         false, false);
        }

        // online softmax update (rows live in 16-lane half groups)
        float rmax[8];
#pragma unroll
        for (int j = 0; j < 8; ++j) {
            s0[j] += mb0;
            s1[j] += mb1;
            float r = fmaxf(s0[j], s1[j]);
            r = fmaxf(r, __shfl_xor(r, 1));
            r = fmaxf(r, __shfl_xor(r, 2));
            r = fmaxf(r, __shfl_xor(r, 4));
            r = fmaxf(r, __shfl_xor(r, 8));
            rmax[j] = r;
        }
#pragma unroll
        for (int j = 0; j < 8; ++j) {
            float mn    = fmaxf(mrow[j], rmax[j]);
            float alpha = __expf(mrow[j] - mn);
            mrow[j]     = mn;
            float p0    = __expf(s0[j] - mn);
            float p1    = __expf(s1[j] - mn);
            float rs    = p0 + p1;
            rs += __shfl_xor(rs, 1);
            rs += __shfl_xor(rs, 2);
            rs += __shfl_xor(rs, 4);
            rs += __shfl_xor(rs, 8);
            lrow[j] = lrow[j] * alpha + rs;
#pragma unroll
            for (int t = 0; t < 16; ++t) acc[t][j] *= alpha;
            const int m = j + half * 8;
            Pt[wave][m][l15]      = f2bf(p0);   // transpose P via per-wave LDS
            Pt[wave][m][16 + l15] = f2bf(p1);
        }

        // ctx += P @ V   (P is 16x32 A-operand; V from transposed LDS tile)
        BFrag pf;
        pf.q[0] = *(const uint4*)&Pt[wave][l15][half * 8];
        pf.q[1] = *(const uint4*)&Pt[wave][l15][16 + half * 8];
#pragma unroll
        for (int dt = 0; dt < 16; ++dt) {
            BFrag vf;
            vf.q[0] = *(const uint4*)&Vt[dt * 16 + l15][half * 8];
            vf.q[1] = *(const uint4*)&Vt[dt * 16 + l15][16 + half * 8];
            acc[dt] = __builtin_amdgcn_wmma_f32_16x16x32_bf16(
                false, pf.v, false, vf.v, (short)0, acc[dt], false, false);
        }
        __syncthreads();
    }

    float inv[8];
#pragma unroll
    for (int j = 0; j < 8; ++j) inv[j] = 1.0f / lrow[j];
#pragma unroll
    for (int dt = 0; dt < 16; ++dt) {
#pragma unroll
        for (int j = 0; j < 8; ++j) {
            const int m   = j + half * 8;
            const int row = qbase + m;
            Ctx[(size_t)row * DMODEL + doff + dt * 16 + l15] =
                f2bf(acc[dt][j] * inv[j]);
        }
    }
}

// ---------------------------------------------------------------------------
extern "C" void kernel_launch(void* const* d_in, const int* in_sizes, int n_in,
                              void* d_out, int out_size, void* d_ws,
                              size_t ws_size, hipStream_t stream) {
    (void)in_sizes; (void)n_in; (void)out_size; (void)ws_size;
    const float* q  = (const float*)d_in[0];
    const float* k  = (const float*)d_in[1];
    const float* v  = (const float*)d_in[2];
    const float* w1 = (const float*)d_in[3];
    const float* b1 = (const float*)d_in[4];
    const float* w2 = (const float*)d_in[5];
    const float* b2 = (const float*)d_in[6];
    const float* wq = (const float*)d_in[7];
    const float* bq = (const float*)d_in[8];
    const float* wk = (const float*)d_in[9];
    const float* bk = (const float*)d_in[10];
    const float* wv = (const float*)d_in[11];
    const float* bv = (const float*)d_in[12];
    const float* wo = (const float*)d_in[13];
    const float* bo = (const float*)d_in[14];

    // workspace carve-up (~67 MB total)
    char* ws = (char*)d_ws;
    auto alloc = [&](size_t bytes) -> char* {
        char* p = ws;
        ws += (bytes + 255) & ~(size_t)255;
        return p;
    };
    const size_t ND = (size_t)NTOK * DMODEL;
    unsigned short* qbf  = (unsigned short*)alloc(ND * 2);
    unsigned short* kbf  = (unsigned short*)alloc(ND * 2);
    unsigned short* vbf  = (unsigned short*)alloc(ND * 2);
    unsigned short* w1bf = (unsigned short*)alloc((size_t)DMODEL * 256 * 2);
    unsigned short* wqbf = (unsigned short*)alloc((size_t)DMODEL * DMODEL * 2);
    unsigned short* wkbf = (unsigned short*)alloc((size_t)DMODEL * DMODEL * 2);
    unsigned short* wvbf = (unsigned short*)alloc((size_t)DMODEL * DMODEL * 2);
    unsigned short* wobf = (unsigned short*)alloc((size_t)DMODEL * DMODEL * 2);
    unsigned short* hbf  = (unsigned short*)alloc((size_t)NTOK * 256 * 2);
    unsigned short* qp   = (unsigned short*)alloc(ND * 2);
    unsigned short* kp   = (unsigned short*)alloc(ND * 2);
    unsigned short* vp   = (unsigned short*)alloc(ND * 2);
    unsigned short* ctx  = (unsigned short*)alloc(ND * 2);
    float* maskbias      = (float*)alloc((size_t)NTOK * 4);
    float* maskmul       = (float*)alloc((size_t)NTOK * 4);

    // 1) fp32 -> bf16
    cvt_kernel<<<2048, 256, 0, stream>>>(q, qbf, (int)ND);
    cvt_kernel<<<2048, 256, 0, stream>>>(k, kbf, (int)ND);
    cvt_kernel<<<2048, 256, 0, stream>>>(v, vbf, (int)ND);
    cvt_kernel<<<512, 256, 0, stream>>>(w1, w1bf, DMODEL * 256);
    cvt_kernel<<<2048, 256, 0, stream>>>(wq, wqbf, DMODEL * DMODEL);
    cvt_kernel<<<2048, 256, 0, stream>>>(wk, wkbf, DMODEL * DMODEL);
    cvt_kernel<<<2048, 256, 0, stream>>>(wv, wvbf, DMODEL * DMODEL);
    cvt_kernel<<<2048, 256, 0, stream>>>(wo, wobf, DMODEL * DMODEL);

    // 2) sparsity predictor: h = relu(q @ w1 + b1), then scores -> masks
    gemm_kernel<1><<<dim3(NTOK / 64, 256 / 16), 128, 0, stream>>>(
        qbf, w1bf, b1, hbf, nullptr, NTOK, 256, DMODEL, 1.0f);
    score_kernel<<<NTOK / 256, 256, 0, stream>>>(hbf, w2, b2, maskbias, maskmul,
                                                 NTOK);

    // 3) QKV projections (Q pre-scaled by 1/sqrt(HDIM) = 1/16)
    gemm_kernel<0><<<dim3(NTOK / 64, DMODEL / 16), 128, 0, stream>>>(
        qbf, wqbf, bq, qp, nullptr, NTOK, DMODEL, DMODEL, 0.0625f);
    gemm_kernel<0><<<dim3(NTOK / 64, DMODEL / 16), 128, 0, stream>>>(
        kbf, wkbf, bk, kp, nullptr, NTOK, DMODEL, DMODEL, 1.0f);
    gemm_kernel<0><<<dim3(NTOK / 64, DMODEL / 16), 128, 0, stream>>>(
        vbf, wvbf, bv, vp, nullptr, NTOK, DMODEL, DMODEL, 1.0f);

    // 4) flash attention over masked keys (K tile via TDM)
    attn_kernel<<<dim3(NTOK / 64, NHEAD), 128, 0, stream>>>(qp, kp, vp, maskbias,
                                                            ctx, NTOK);

    // 5) output projection + scatter mask -> fp32 d_out
    gemm_kernel<2><<<dim3(NTOK / 64, DMODEL / 16), 128, 0, stream>>>(
        ctx, wobf, bo, d_out, maskmul, NTOK, DMODEL, DMODEL, 1.0f);
}
